// MultLayer3_15522011808177
// MI455X (gfx1250) — matrix operand order repaired and checked by reading, not compiled, not run on hardware
//
#include <hip/hip_runtime.h>

// MI455X / gfx1250: wave32, WMMA bf16 16x16x32 path.
typedef __attribute__((ext_vector_type(16))) __bf16 v16bf;
typedef __attribute__((ext_vector_type(8)))  __bf16 v8bf;
typedef __attribute__((ext_vector_type(8)))  float  v8f;

#define B_   16
#define CIN  64
#define COUT 96
#define IH_  128
#define IW_  128
#define PH   130   // padded height (pad=1 each side)
#define PW   130   // padded width

__device__ __forceinline__ unsigned short f2bf(float f) {
  union { float f; unsigned u; } v; v.f = f;
  unsigned r = v.u + 0x7fffu + ((v.u >> 16) & 1u);  // round-to-nearest-even
  return (unsigned short)(r >> 16);
}

// ---------------------------------------------------------------------------
// Prep 1: x (fp32 NCHW) -> xt (bf16, padded, NHWC: [b][y][x][c], c contiguous)
// lane == x  => reads of x are 128B-coalesced per channel iteration;
// each lane writes its own 128B pixel via 8x b128 stores (L2 merges).
// ---------------------------------------------------------------------------
__global__ void xform_input(const float* __restrict__ x,
                            unsigned short* __restrict__ xt) {
  int pid = blockIdx.x * blockDim.x + threadIdx.x;
  const int NPIX = B_ * PH * PW;
  if (pid >= NPIX) return;
  int xx = pid % PW;
  int yy = (pid / PW) % PH;
  int b  = pid / (PW * PH);

  unsigned dw[CIN / 2];
  bool pad = (yy == 0) || (yy == PH - 1) || (xx == 0) || (xx == PW - 1);
  if (pad) {
#pragma unroll
    for (int k = 0; k < CIN / 2; ++k) dw[k] = 0u;
  } else {
    const float* src = x + (size_t)b * CIN * IH_ * IW_
                         + (size_t)(yy - 1) * IW_ + (xx - 1);
#pragma unroll
    for (int k = 0; k < CIN / 2; ++k) {
      float f0 = src[(size_t)(2 * k)     * IH_ * IW_];
      float f1 = src[(size_t)(2 * k + 1) * IH_ * IW_];
      dw[k] = (unsigned)f2bf(f0) | ((unsigned)f2bf(f1) << 16);
    }
  }
  uint4* dst = (uint4*)(xt + (size_t)pid * CIN);
#pragma unroll
  for (int k = 0; k < CIN / 8; ++k) {
    uint4 t; t.x = dw[4*k]; t.y = dw[4*k+1]; t.z = dw[4*k+2]; t.w = dw[4*k+3];
    dst[k] = t;
  }
}

// ---------------------------------------------------------------------------
// Prep 2: effective weights.
// Whats[pq][o][k], k = (rr*4+cc)*64 + c   (rr = y-offset, cc = x-offset)
// Ŵ[p,q,rr,cc] = Σ_{u,v} (IHg[p,u]·Hg[cc,u]) (IHg[q,v]·Hg[rr,v]) w[o,c,u,v]
// with Hg = row-permuted Hadamard per group, IHg = rows 1,2 of Hg.
// (The reference's patch extraction transposes u<->v; folded in: cc couples
//  to p, rr couples to q.)
// ---------------------------------------------------------------------------
__global__ void make_weights(const float* __restrict__ w,
                             unsigned short* __restrict__ wh) {
  int t = blockIdx.x * blockDim.x + threadIdx.x;
  const int NT = 4 * COUT * CIN;
  if (t >= NT) return;
  int c  = t % CIN;
  int o  = (t / CIN) % COUT;
  int pq = t / (CIN * COUT);
  int p = pq >> 1, q = pq & 1;
  int g = o >> 5;  // 32 output channels per group

  const int Hm[4][4]   = {{1,1,1,1},{1,-1,1,-1},{1,1,-1,-1},{1,-1,-1,1}};
  const int perm[3][4] = {{0,1,2,3},{1,0,2,3},{3,1,2,0}};  // H, H2, H3 rows

  float wv[4][4];
#pragma unroll
  for (int u = 0; u < 4; ++u)
#pragma unroll
    for (int v = 0; v < 4; ++v)
      wv[u][v] = w[((size_t)(o * CIN + c) * 4 + u) * 4 + v];

  unsigned short* dst = wh + (size_t)(pq * COUT + o) * 1024 + c;
#pragma unroll
  for (int rr = 0; rr < 4; ++rr) {
#pragma unroll
    for (int cc = 0; cc < 4; ++cc) {
      float acc = 0.f;
#pragma unroll
      for (int u = 0; u < 4; ++u) {
        int mu = Hm[perm[g][p + 1]][u] * Hm[perm[g][cc]][u];
#pragma unroll
        for (int v = 0; v < 4; ++v) {
          int mv = Hm[perm[g][q + 1]][v] * Hm[perm[g][rr]][v];
          acc += (float)(mu * mv) * wv[u][v];
        }
      }
      dst[(rr * 4 + cc) * CIN] = f2bf(acc);
    }
  }
}

// ---------------------------------------------------------------------------
// Main fused kernel: 4 implicit GEMMs (one per 2x2 output sub-position) with
// shared A. Block = 192 threads (6 waves): wave w owns N-tile n0 = 16*w,
// 2 M-tiles of 16 cells along j, all 4 (p,q). 32 K-steps of 32 (tap-major,
// channel fastest). bf16 WMMA, fp32 accumulate.
// ---------------------------------------------------------------------------
__global__ void __launch_bounds__(192)
winograd_main(const unsigned short* __restrict__ xt,
              const unsigned short* __restrict__ wh,
              const float* __restrict__ bias,
              float* __restrict__ out) {
  const int jp   = blockIdx.x;        // 0..1  -> j0 = 32*jp
  const int i    = blockIdx.y;        // 0..63
  const int b    = blockIdx.z;        // 0..15
  const int wv_  = threadIdx.x >> 5;  // wave id 0..5 -> n-tile
  const int lane = threadIdx.x & 31;
  const int h    = lane >> 4;
  const int ln   = lane & 15;
  const int n0   = wv_ * 16;
  const int j0   = jp * 32;

  const v8f zero = {0.f, 0.f, 0.f, 0.f, 0.f, 0.f, 0.f, 0.f};
  v8f acc[2][4];
#pragma unroll
  for (int mt = 0; mt < 2; ++mt)
#pragma unroll
    for (int pq = 0; pq < 4; ++pq) acc[mt][pq] = zero;

  const unsigned short* xbase = xt + (size_t)b * PH * PW * CIN;
  const int co0 = h ? 8 : 0;  // A layout: half-wave K split 0-7/8-15, 16-23/24-31

#pragma unroll 2
  for (int kk = 0; kk < 32; ++kk) {
    const int tap = kk >> 1;
    const int rr = tap >> 2, cc = tap & 3;     // rr = y offset, cc = x offset
    const int chalf = (kk & 1) * 32;

    v16bf A[2];
#pragma unroll
    for (int mt = 0; mt < 2; ++mt) {
      int j = j0 + mt * 16 + ln;               // A row M = output cell
      const unsigned short* pa =
          xbase + ((size_t)(2 * i + rr) * PW + (2 * j + cc)) * CIN + chalf + co0;
      v8bf lo = *(const v8bf*)pa;              // K (8h)..(8h+7)
      v8bf hi = *(const v8bf*)(pa + 16);       // K (16+8h)..(23+8h)
      A[mt] = __builtin_shufflevector(lo, hi, 0,1,2,3,4,5,6,7,8,9,10,11,12,13,14,15);
    }

#pragma unroll
    for (int pq = 0; pq < 4; ++pq) {
      // B layout: lane = N, K 8h..8h+7 then 16+8h..23+8h (matches A)
      const unsigned short* pb =
          wh + (size_t)(pq * COUT + n0 + ln) * 1024 + kk * 32 + 8 * h;
      v8bf blo = *(const v8bf*)pb;
      v8bf bhi = *(const v8bf*)(pb + 16);
      v16bf Bm = __builtin_shufflevector(blo, bhi, 0,1,2,3,4,5,6,7,8,9,10,11,12,13,14,15);
#pragma unroll
      for (int mt = 0; mt < 2; ++mt) {
        acc[mt][pq] = __builtin_amdgcn_wmma_f32_16x16x32_bf16(
            false, A[mt], false, Bm, (short)0, acc[mt][pq], false, false);
      }
    }
  }

  // Epilogue: C/D layout lane -> (M = v + 8h, N = ln). Fuse bias + 2x2
  // sub-pixel interleave; pair q=0/q=1 into float2 stores.
  const float bv = bias[n0 + ln];
  const int o = n0 + ln;
#pragma unroll
  for (int mt = 0; mt < 2; ++mt) {
#pragma unroll
    for (int p = 0; p < 2; ++p) {
      const int y = 2 * i + p;
      float* orow = out + (((size_t)b * COUT + o) * IH_ + y) * IW_;
#pragma unroll
      for (int v = 0; v < 8; ++v) {
        int j = j0 + mt * 16 + v + 8 * h;
        float2 o2;
        o2.x = acc[mt][2 * p + 0][v] + bv;
        o2.y = acc[mt][2 * p + 1][v] + bv;
        *(float2*)(orow + 2 * j) = o2;
      }
    }
  }
}

// ---------------------------------------------------------------------------
extern "C" void kernel_launch(void* const* d_in, const int* in_sizes, int n_in,
                              void* d_out, int out_size, void* d_ws, size_t ws_size,
                              hipStream_t stream) {
  const float* x    = (const float*)d_in[0];
  const float* wgt  = (const float*)d_in[1];
  const float* bias = (const float*)d_in[2];
  float* out = (float*)d_out;

  // workspace: xt (padded NHWC bf16) then Whats
  unsigned short* xt = (unsigned short*)d_ws;
  size_t xt_bytes = (size_t)B_ * PH * PW * CIN * sizeof(unsigned short); // ~34.6 MB
  size_t wh_off = (xt_bytes + 255) & ~(size_t)255;
  unsigned short* wh = (unsigned short*)((char*)d_ws + wh_off);          // 768 KB

  {
    int n = B_ * PH * PW;
    xform_input<<<(n + 255) / 256, 256, 0, stream>>>(x, xt);
  }
  {
    int n = 4 * COUT * CIN;
    make_weights<<<(n + 255) / 256, 256, 0, stream>>>(wgt, wh);
  }
  {
    dim3 grid(2, 64, 16);  // (j-pair, i, b)
    winograd_main<<<grid, 192, 0, stream>>>(xt, wh, bias, out);
  }
}
